// SSM_3186865733719
// MI455X (gfx1250) — matrix operand order repaired and checked by reading, compiled, not run on hardware
//
#include <hip/hip_runtime.h>

// ---------------------------------------------------------------------------
// Mamba2-style SSM block for MI455X (gfx1250, wave32, WMMA).
// fp32 storage, bf16 WMMA compute with fp32 accumulation.
// ---------------------------------------------------------------------------

typedef __attribute__((ext_vector_type(16))) __bf16 bf16x16;
typedef __attribute__((ext_vector_type(8)))  float  f32x8;

#define B_       2
#define S_       2048
#define E_       2048
#define NH       64
#define HD       64
#define NS       64
#define NG       8
#define INTER    4096
#define CONV_K   4
#define CHUNK    256
#define CONV_DIM 5120          // INTER + 2*NG*NS
#define PROJ     9280          // INTER + CONV_DIM + NH
#define TOKENS   (B_ * S_)     // 4096
#define NCHUNK   (S_ / CHUNK)  // 8

union FragCvt { bf16x16 v; float4 q[2]; };

__device__ __forceinline__ f32x8 wmma_bf16(bf16x16 a, bf16x16 b, f32x8 c) {
  // (neg_a, A, neg_b, B, c_mod, C, reuse_a, reuse_b)
  return __builtin_amdgcn_wmma_f32_16x16x32_bf16(false, a, false, b,
                                                 (short)0, c, false, false);
}

// ---------------------------------------------------------------------------
// Generic fp32-in / fp32-out GEMM, bf16 WMMA compute.
// Block tile 128x128, K-step 32, 8 waves each computing a 32x64 sub-tile.
// A-fragment (16x32 bf16): lane m = lane&15, half = lane>>4;
//   elems 0..7  <-> K = half*8 + e
//   elems 8..15 <-> K = 16 + half*8 + (e-8)
// B-fragment: lane n = lane&15, half = lane>>4; same K mapping as A-fragment.
// C/D: VGPR e -> M = e + 8*(lane>>4), N = lane&15.
// ---------------------------------------------------------------------------
__global__ __launch_bounds__(256) void wmma_gemm_kernel(
    const float* __restrict__ A, const float* __restrict__ Bm,
    float* __restrict__ C, int M, int N, int K)
{
  __shared__ __bf16 sA[128][32];   // [m][k]
  __shared__ __bf16 sBt[128][32];  // [n][k]  (B transposed)

  const int tid  = threadIdx.x;
  const int lane = tid & 31;
  const int wave = tid >> 5;
  const int ln   = lane & 15;
  const int kh   = lane >> 4;

  const int m0 = blockIdx.y * 128;
  const int n0 = blockIdx.x * 128;

  const int mbase = (wave >> 1) * 32;  // 4 waves along M
  const int nbase = (wave & 1) * 64;   // 2 waves along N

  f32x8 acc[2][4];
  #pragma unroll
  for (int i = 0; i < 2; ++i)
    #pragma unroll
    for (int j = 0; j < 4; ++j) acc[i][j] = {};

  const int rowA = tid >> 1;        // 0..127
  const int cgA  = (tid & 1) * 16;  // 0 / 16

  for (int k0 = 0; k0 < K; k0 += 32) {
    // Stage A tile (128x32) -> bf16 LDS
    {
      const float* ap = A + (size_t)(m0 + rowA) * K + k0 + cgA;
      if (k0 + 32 < K) __builtin_prefetch(ap + 32, 0, 1);
      #pragma unroll
      for (int j = 0; j < 16; ++j) sA[rowA][cgA + j] = (__bf16)ap[j];
    }
    // Stage B tile transposed (n-major) -> bf16 LDS
    {
      const int n = n0 + rowA;
      if (n < N) {
        #pragma unroll
        for (int j = 0; j < 16; ++j)
          sBt[rowA][cgA + j] = (__bf16)Bm[(size_t)(k0 + cgA + j) * N + n];
      } else {
        #pragma unroll
        for (int j = 0; j < 16; ++j) sBt[rowA][cgA + j] = (__bf16)0.f;
      }
    }
    __syncthreads();

    FragCvt fa[2], fb[4];
    #pragma unroll
    for (int i = 0; i < 2; ++i) {
      const int m = mbase + i * 16 + ln;
      fa[i].q[0] = *(const float4*)&sA[m][kh * 8];
      fa[i].q[1] = *(const float4*)&sA[m][16 + kh * 8];
    }
    #pragma unroll
    for (int j = 0; j < 4; ++j) {
      const int n = nbase + j * 16 + ln;
      fb[j].q[0] = *(const float4*)&sBt[n][kh * 8];
      fb[j].q[1] = *(const float4*)&sBt[n][16 + kh * 8];
    }
    #pragma unroll
    for (int i = 0; i < 2; ++i)
      #pragma unroll
      for (int j = 0; j < 4; ++j)
        acc[i][j] = wmma_bf16(fa[i].v, fb[j].v, acc[i][j]);

    __syncthreads();
  }

  #pragma unroll
  for (int i = 0; i < 2; ++i)
    #pragma unroll
    for (int j = 0; j < 4; ++j) {
      const int col = n0 + nbase + j * 16 + ln;
      if (col < N) {
        #pragma unroll
        for (int e = 0; e < 8; ++e) {
          const int row = m0 + mbase + i * 16 + e + 8 * kh;
          C[(size_t)row * N + col] = acc[i][j][e];
        }
      }
    }
}

// ---------------------------------------------------------------------------
// Depthwise causal conv1d (K=4) + bias + SiLU over proj[:, INTER:INTER+CONV_DIM]
// ---------------------------------------------------------------------------
__global__ __launch_bounds__(256) void conv_silu_kernel(
    const float* __restrict__ proj, const float* __restrict__ w,
    const float* __restrict__ bias, float* __restrict__ out)
{
  const size_t idx = (size_t)blockIdx.x * 256 + threadIdx.x;
  if (idx >= (size_t)TOKENS * CONV_DIM) return;
  const int cdim = (int)(idx % CONV_DIM);
  const int t    = (int)(idx / CONV_DIM);
  const int s    = t % S_;
  float acc = bias[cdim];
  #pragma unroll
  for (int k = 0; k < CONV_K; ++k) {
    const int sp = s - (CONV_K - 1) + k;
    if (sp >= 0)
      acc += proj[(size_t)(t - (CONV_K - 1) + k) * PROJ + INTER + cdim]
             * w[cdim * CONV_K + k];
  }
  out[idx] = acc / (1.f + __expf(-acc));   // SiLU
}

// ---------------------------------------------------------------------------
// dt = clip(softplus(proj[..., -NH:] + dt_bias), >= 0)
// ---------------------------------------------------------------------------
__global__ __launch_bounds__(256) void dt_softplus_kernel(
    const float* __restrict__ proj, const float* __restrict__ dt_bias,
    float* __restrict__ out)
{
  const int idx = blockIdx.x * 256 + threadIdx.x;
  if (idx >= TOKENS * NH) return;
  const int hh = idx & (NH - 1);
  const int t  = idx >> 6;
  const float x  = proj[(size_t)t * PROJ + INTER + CONV_DIM + hh] + dt_bias[hh];
  const float sp = (x > 20.f) ? x : log1pf(__expf(x));
  out[idx] = sp > 0.f ? sp : 0.f;
}

// ---------------------------------------------------------------------------
// Per-(batch, chunk, head) SSM core. Chunks are independent in this reference
// (per-chunk state is only applied within the same chunk).
//   G  = C (256x64) @ B^T (64x256)          -> WMMA, K=NS=64
//   Yd = (L .* G) (256x256) @ Hs (256x64)   -> WMMA, K=s-blocks of 64
//   Yoff[l,d] = C[l,d] * state[d] * exp(Acum[l]);  state = sum_s B*decay*Hs
// One workgroup (8 waves) per (b,c,h); each wave owns two 16-row stripes.
// ---------------------------------------------------------------------------
__global__ __launch_bounds__(256) void ssm_chunk_kernel(
    const float* __restrict__ convp,  // (TOKENS, CONV_DIM), post conv+SiLU
    const float* __restrict__ dtp,    // (TOKENS, NH)
    const float* __restrict__ A_log,  // (NH,)
    float* __restrict__ ybuf)         // (TOKENS, INTER) = Yd + Yoff
{
  __shared__ __bf16 sHsT[64][256];    // Hs transposed: [d][l], bf16 (32 KB)
  __shared__ __bf16 sG[8][16][64];    // per-wave masked-G relayout scratch
  __shared__ float  sAc[256];         // inclusive cumsum of A*dt
  __shared__ float  sdt[256];
  __shared__ float  sState[64];

  const int tid  = threadIdx.x;
  const int lane = tid & 31;
  const int wave = tid >> 5;
  const int ln   = lane & 15;
  const int kh   = lane >> 4;

  const int bid  = blockIdx.x;
  const int b    = bid / (NCHUNK * NH);
  const int rem  = bid % (NCHUNK * NH);
  const int cch  = rem / NH;
  const int h    = rem % NH;
  const int g    = h / (NH / NG);
  const int tok0 = b * S_ + cch * CHUNK;

  // dt and A*dt per chunk position
  {
    const float dtv = dtp[(size_t)(tok0 + tid) * NH + h];
    sdt[tid] = dtv;
    sAc[tid] = -__expf(A_log[h]) * dtv;
  }
  __syncthreads();
  if (tid == 0) {
    float run = 0.f;
    for (int i = 0; i < CHUNK; ++i) { run += sAc[i]; sAc[i] = run; }
  }
  __syncthreads();

  // Hs^T = (hidden * dt)^T as bf16 (B operand of the 2nd GEMM)
  for (int r = 0; r < 64; ++r) {
    const int idx = r * 256 + tid;
    const int l = idx >> 6;
    const int d = idx & 63;
    const float hv = convp[(size_t)(tok0 + l) * CONV_DIM + h * HD + d];
    sHsT[d][l] = (__bf16)(hv * sdt[l]);
  }
  __syncthreads();

  // chunk-local state[n] = sum_s B[s,n] * exp(Acum[255]-Acum[s]) * Hs[s,n]
  if (tid < 64) {
    const int n = tid;
    const float atot = sAc[CHUNK - 1];
    float acc = 0.f;
    for (int s = 0; s < CHUNK; ++s) {
      const float bv = convp[(size_t)(tok0 + s) * CONV_DIM + INTER + g * NS + n];
      acc += bv * __expf(atot - sAc[s]) * (float)sHsT[n][s];
    }
    sState[n] = acc;
  }
  __syncthreads();

  for (int rb = 0; rb < 2; ++rb) {
    const int l0 = rb * 128 + wave * 16;

    // A fragments of C-matrix rows (16 x 64, two K=32 chunks), from global
    bf16x16 faC[2];
    #pragma unroll
    for (int ka = 0; ka < 2; ++ka) {
      const float* p = convp + (size_t)(tok0 + l0 + ln) * CONV_DIM
                       + INTER + NG * NS + g * NS + ka * 32;
      #pragma unroll
      for (int e = 0; e < 8; ++e) faC[ka][e] = (__bf16)p[kh * 16 + e];
      #pragma unroll
      for (int e = 0; e < 8; ++e) faC[ka][8 + e] = (__bf16)p[kh * 16 + 8 + e];
    }

    f32x8 yacc[4];
    #pragma unroll
    for (int jn = 0; jn < 4; ++jn) yacc[jn] = {};

    for (int sb = 0; sb < 4; ++sb) {
      const int sbb = sb * 64;

      // G stripe (16 x 64) = C @ B^T, then causal decay mask -> sG (bf16)
      #pragma unroll
      for (int js = 0; js < 4; ++js) {
        f32x8 gt = {};
        #pragma unroll
        for (int ka = 0; ka < 2; ++ka) {
          bf16x16 fbB;
          const float* p = convp
              + (size_t)(tok0 + sbb + js * 16 + ln) * CONV_DIM
              + INTER + g * NS + ka * 32 + kh * 16;
          #pragma unroll
          for (int e = 0; e < 16; ++e) fbB[e] = (__bf16)p[e];
          gt = wmma_bf16(faC[ka], fbB, gt);
        }
        #pragma unroll
        for (int e = 0; e < 8; ++e) {
          const int lrow = l0 + e + 8 * kh;
          const int scol = sbb + js * 16 + ln;
          const float m = (lrow >= scol)
              ? gt[e] * __expf(sAc[lrow] - sAc[scol]) : 0.f;
          sG[wave][e + 8 * kh][js * 16 + ln] = (__bf16)m;
        }
      }
      __syncthreads();

      // Y += Gm (16x64) @ Hs (64x64 stripe)
      #pragma unroll
      for (int ka = 0; ka < 2; ++ka) {
        FragCvt faG;
        faG.q[0] = *(const float4*)&sG[wave][ln][ka * 32 + kh * 8];
        faG.q[1] = *(const float4*)&sG[wave][ln][ka * 32 + 16 + kh * 8];
        #pragma unroll
        for (int jn = 0; jn < 4; ++jn) {
          FragCvt fbH;
          const int d  = jn * 16 + ln;
          const int kb = sbb + ka * 32 + kh * 8;
          fbH.q[0] = *(const float4*)&sHsT[d][kb];
          fbH.q[1] = *(const float4*)&sHsT[d][kb + 16];
          yacc[jn] = wmma_bf16(faG.v, fbH.v, yacc[jn]);
        }
      }
      __syncthreads();
    }

    // epilogue: add Yoff, store
    #pragma unroll
    for (int jn = 0; jn < 4; ++jn) {
      const int d = jn * 16 + ln;
      #pragma unroll
      for (int e = 0; e < 8; ++e) {
        const int lrow = l0 + e + 8 * kh;
        const float cf = convp[(size_t)(tok0 + lrow) * CONV_DIM
                               + INTER + NG * NS + g * NS + d];
        const float yv = yacc[jn][e] + cf * sState[d] * __expf(sAc[lrow]);
        ybuf[(size_t)(tok0 + lrow) * INTER + h * HD + d] = yv;
      }
    }
  }
}

// ---------------------------------------------------------------------------
// y = norm_w * (yg * rsqrt(mean(yg^2)+eps)), yg = (y + D*H) * silu(gate)
// One workgroup per token; reduction over INTER=4096.
// ---------------------------------------------------------------------------
__global__ __launch_bounds__(256) void gated_norm_kernel(
    float* __restrict__ y, const float* __restrict__ convp,
    const float* __restrict__ proj, const float* __restrict__ Dv,
    const float* __restrict__ normw)
{
  __shared__ float red[256];
  const int t   = blockIdx.x;
  const int tid = threadIdx.x;
  float ss = 0.f;
  #pragma unroll
  for (int j = 0; j < INTER / 256; ++j) {
    const int i = tid + j * 256;
    float yv = y[(size_t)t * INTER + i];
    yv += Dv[i >> 6] * convp[(size_t)t * CONV_DIM + i];   // D residual
    const float gv = proj[(size_t)t * PROJ + i];          // gate
    const float yg = yv * (gv / (1.f + __expf(-gv)));
    y[(size_t)t * INTER + i] = yg;
    ss += yg * yg;
  }
  red[tid] = ss;
  __syncthreads();
  for (int off = 128; off > 0; off >>= 1) {
    if (tid < off) red[tid] += red[tid + off];
    __syncthreads();
  }
  const float rs = rsqrtf(red[0] / (float)INTER + 1e-6f);
  #pragma unroll
  for (int j = 0; j < INTER / 256; ++j) {
    const int i = tid + j * 256;
    y[(size_t)t * INTER + i] = normw[i] * y[(size_t)t * INTER + i] * rs;
  }
}

// ---------------------------------------------------------------------------
extern "C" void kernel_launch(void* const* d_in, const int* in_sizes, int n_in,
                              void* d_out, int out_size, void* d_ws, size_t ws_size,
                              hipStream_t stream) {
  const float* x      = (const float*)d_in[0];
  // d_in[1] = mask (unused by reference)
  const float* W_in   = (const float*)d_in[2];
  const float* conv_w = (const float*)d_in[3];
  const float* conv_b = (const float*)d_in[4];
  const float* dt_b   = (const float*)d_in[5];
  const float* A_log  = (const float*)d_in[6];
  const float* Dv     = (const float*)d_in[7];
  const float* norm_w = (const float*)d_in[8];
  const float* W_out  = (const float*)d_in[9];
  float* out = (float*)d_out;

  // workspace layout (fp32): proj | conv | dt | y   (~304 MB total)
  float* proj = (float*)d_ws;
  float* conv = proj + (size_t)TOKENS * PROJ;
  float* dtb  = conv + (size_t)TOKENS * CONV_DIM;
  float* ybuf = dtb  + (size_t)TOKENS * NH;

  // 1) proj = X @ W_in            (4096 x 2048) @ (2048 x 9280)
  wmma_gemm_kernel<<<dim3((PROJ + 127) / 128, TOKENS / 128), 256, 0, stream>>>(
      x, W_in, proj, TOKENS, PROJ, E_);

  // 2) depthwise causal conv + SiLU
  conv_silu_kernel<<<(int)(((size_t)TOKENS * CONV_DIM + 255) / 256), 256, 0,
                    stream>>>(proj, conv_w, conv_b, conv);

  // 3) dt = softplus(dt_raw + dt_bias)
  dt_softplus_kernel<<<(TOKENS * NH + 255) / 256, 256, 0, stream>>>(
      proj, dt_b, dtb);

  // 4) chunked SSM core (WMMA), one block per (b, chunk, head)
  ssm_chunk_kernel<<<B_ * NCHUNK * NH, 256, 0, stream>>>(conv, dtb, A_log, ybuf);

  // 5) gated RMSNorm (in-place on ybuf)
  gated_norm_kernel<<<TOKENS, 256, 0, stream>>>(ybuf, conv, proj, Dv, norm_w);

  // 6) out = y_norm @ W_out       (4096 x 4096) @ (4096 x 2048)
  wmma_gemm_kernel<<<dim3(E_ / 128, TOKENS / 128), 256, 0, stream>>>(
      ybuf, W_out, out, TOKENS, E_, INTER);
}
